// MyMultiheadAttention_13907104104511
// MI455X (gfx1250) — matrix operand order repaired and checked by reading, not compiled, not run on hardware
//
#include <hip/hip_runtime.h>
#include <cmath>

// Problem constants (match reference)
#define BB 2
#define SS 2048
#define DD 1024
#define HH 16
#define DKK 64

typedef _Float16 h16 __attribute__((ext_vector_type(16)));
typedef float    f8  __attribute__((ext_vector_type(8)));

__device__ __forceinline__ f8 wmma_f16(h16 a, h16 b, f8 c) {
  // D = A(16x32 f16) * B(32x16 f16) + C(16x16 f32)
  return __builtin_amdgcn_wmma_f32_16x16x32_f16(false, a, false, b, (short)0, c,
                                                false, false);
}

// ---- fragment loaders (layouts per CDNA5 ISA 7.12.2, wave32) -------------
// A 16x32 f16: lanes 0-15: M=lane, K={k0..k0+7, k0+16..k0+23}
//              lanes 16-31: M=lane-16, K={k0+8..k0+15, k0+24..k0+31}
__device__ __forceinline__ h16 a_frag_f32(const float* X, int ld, int m0, int k0, int lane) {
  int l = lane & 15, hi = lane >> 4;
  const float* p = X + (size_t)(m0 + l) * ld + k0 + (hi ? 8 : 0);
  h16 a;
#pragma unroll
  for (int i = 0; i < 8; ++i) a[i] = (_Float16)p[i];
#pragma unroll
  for (int i = 0; i < 8; ++i) a[8 + i] = (_Float16)p[16 + i];
  return a;
}

__device__ __forceinline__ h16 a_frag_f16(const _Float16* X, int ld, int m0, int k0, int lane) {
  int l = lane & 15, hi = lane >> 4;
  const _Float16* p = X + (size_t)(m0 + l) * ld + k0 + (hi ? 8 : 0);
  h16 a;
#pragma unroll
  for (int i = 0; i < 8; ++i) a[i] = p[i];
#pragma unroll
  for (int i = 0; i < 8; ++i) a[8 + i] = p[16 + i];
  return a;
}

// B 32x16 f16: lanes 0-15: N=lane, K={k0..k0+7, k0+16..k0+23} ; lanes 16-31: N=lane-16, K={k0+8..k0+15, k0+24..k0+31}.
// Source Wn is row-major [N][K] (each lane's K-chunk contiguous).
__device__ __forceinline__ h16 b_frag_f32(const float* Wn, int ld, int n0, int k0, int lane) {
  int l = lane & 15, hi = lane >> 4;
  const float* p = Wn + (size_t)(n0 + l) * ld + k0 + (hi ? 8 : 0);
  h16 b;
#pragma unroll
  for (int i = 0; i < 8; ++i) b[i] = (_Float16)p[i];
#pragma unroll
  for (int i = 0; i < 8; ++i) b[8 + i] = (_Float16)p[16 + i];
  return b;
}

__device__ __forceinline__ h16 b_frag_f16(const _Float16* Wn, int ld, int n0, int k0, int lane) {
  int l = lane & 15, hi = lane >> 4;
  const _Float16* p = Wn + (size_t)(n0 + l) * ld + k0 + (hi ? 8 : 0);
  h16 b;
#pragma unroll
  for (int i = 0; i < 8; ++i) b[i] = p[i];
#pragma unroll
  for (int i = 0; i < 8; ++i) b[8 + i] = p[16 + i];
  return b;
}

// -------------------- kernel 1: fused QKV projections ---------------------
// grid.y = 0/1/2 -> q/k/v. One wave per 16x16 output tile of X @ W^T.
// q -> Qp[b,h,s,dk] * (1/sqrt(dk));  k -> Kp[b,h,s,dk];  v -> Vt[b,h,dk,s].
__global__ void __launch_bounds__(256)
proj_kernel(const float* __restrict__ Q, const float* __restrict__ K,
            const float* __restrict__ V, const float* __restrict__ Wq,
            const float* __restrict__ Wk, const float* __restrict__ Wv,
            _Float16* __restrict__ Qp, _Float16* __restrict__ Kp,
            _Float16* __restrict__ Vt) {
  int proj = blockIdx.y;
  int wave = blockIdx.x * (blockDim.x >> 5) + (threadIdx.x >> 5);
  int lane = threadIdx.x & 31;
  int tm = wave >> 6, tn = wave & 63;  // 256 M-tiles x 64 N-tiles
  int m0 = tm * 16, n0 = tn * 16;
  const float* X = (proj == 0) ? Q : ((proj == 1) ? K : V);
  const float* W = (proj == 0) ? Wq : ((proj == 1) ? Wk : Wv);

  f8 acc = (f8)0.0f;
  for (int k = 0; k < DD; k += 32)
    acc = wmma_f16(a_frag_f32(X, DD, m0, k, lane), b_frag_f32(W, DD, n0, k, lane), acc);

  int l = lane & 15, hi = lane >> 4;
  int col = n0 + l;
  int h = col >> 6, dk = col & (DKK - 1);
#pragma unroll
  for (int r = 0; r < 8; ++r) {
    int row = m0 + r + (hi ? 8 : 0);       // row = b*S + s
    int b = row >> 11, s = row & (SS - 1);
    float v = acc[r];
    if (proj == 0)
      Qp[(((size_t)(b * HH + h)) * SS + s) * DKK + dk] = (_Float16)(v * 0.125f);
    else if (proj == 1)
      Kp[(((size_t)(b * HH + h)) * SS + s) * DKK + dk] = (_Float16)v;
    else
      Vt[(((size_t)(b * HH + h)) * DKK + dk) * SS + s] = (_Float16)v;
  }
}

// -------------------- kernel 2: flash attention per (b,h) -----------------
// One wave (block of 32) per 16-query strip; stream keys 32 at a time.
__global__ void __launch_bounds__(32)
attn_kernel(const _Float16* __restrict__ Qp, const _Float16* __restrict__ Kp,
            const _Float16* __restrict__ Vt, const int* __restrict__ mask,
            _Float16* __restrict__ Ctx) {
  __shared__ __align__(16) _Float16 plds[16 * 32];  // P tile staging (C->A relayout)

  int lane = threadIdx.x & 31;
  int qt = blockIdx.x & 127;       // S/16 = 128 query tiles
  int bh = blockIdx.x >> 7;        // b*H + h
  int b = bh >> 4, h = bh & 15;
  int q0 = qt * 16;
  const _Float16* Qh = Qp + (size_t)bh * SS * DKK;  // [s][dk]
  const _Float16* Kh = Kp + (size_t)bh * SS * DKK;  // [s][dk]
  const _Float16* Vh = Vt + (size_t)bh * DKK * SS;  // [dk][s]
  int l = lane & 15, hi = lane >> 4;

  // q A-fragments (16 rows x dk 0..63), loaded once
  h16 qa0 = a_frag_f16(Qh, DKK, q0, 0, lane);
  h16 qa1 = a_frag_f16(Qh, DKK, q0, 32, lane);

  f8 o0 = (f8)0.0f, o1 = (f8)0.0f, o2 = (f8)0.0f, o3 = (f8)0.0f;
  float m_r[8], l_r[8];
#pragma unroll
  for (int r = 0; r < 8; ++r) { m_r[r] = -INFINITY; l_r[r] = 0.0f; }

  for (int kk = 0; kk < SS; kk += 32) {
    // scores: q(16x64) @ k^T(64x32) -> two 16x16 tiles (q already scaled 1/8)
    f8 s0 = (f8)0.0f, s1 = (f8)0.0f;
    s0 = wmma_f16(qa0, b_frag_f16(Kh, DKK, kk, 0, lane), s0);
    s0 = wmma_f16(qa1, b_frag_f16(Kh, DKK, kk, 32, lane), s0);
    s1 = wmma_f16(qa0, b_frag_f16(Kh, DKK, kk + 16, 0, lane), s1);
    s1 = wmma_f16(qa1, b_frag_f16(Kh, DKK, kk + 16, 32, lane), s1);

    // faithful masking: masked positions get 1e-9 BEFORE softmax
    int c0 = kk + l, c1 = c0 + 16;
#pragma unroll
    for (int r = 0; r < 8; ++r) {
      int qrow = q0 + r + (hi ? 8 : 0);
      const int* mrow = mask + ((size_t)b * SS + qrow) * SS;
      if (mrow[c0] == 0) s0[r] = 1e-9f;
      if (mrow[c1] == 0) s1[r] = 1e-9f;
    }

    // online softmax: row r of tile lives across the 16-lane half-group
#pragma unroll
    for (int r = 0; r < 8; ++r) {
      float cm = fmaxf(s0[r], s1[r]);
#pragma unroll
      for (int w = 1; w < 16; w <<= 1) cm = fmaxf(cm, __shfl_xor(cm, w));
      float mn = fmaxf(m_r[r], cm);
      float alpha = __expf(m_r[r] - mn);
      float p0 = __expf(s0[r] - mn);
      float p1 = __expf(s1[r] - mn);
      s0[r] = p0; s1[r] = p1;
      float rs = p0 + p1;
#pragma unroll
      for (int w = 1; w < 16; w <<= 1) rs += __shfl_xor(rs, w);
      l_r[r] = l_r[r] * alpha + rs;
      m_r[r] = mn;
      o0[r] *= alpha; o1[r] *= alpha; o2[r] *= alpha; o3[r] *= alpha;
    }

    // P: C-layout -> LDS -> A-layout
#pragma unroll
    for (int r = 0; r < 8; ++r) {
      int row = r + (hi ? 8 : 0);
      plds[row * 32 + l] = (_Float16)s0[r];
      plds[row * 32 + l + 16] = (_Float16)s1[r];
    }
    __syncthreads();
    h16 pa;
    {
      const _Float16* pp = plds + l * 32 + (hi ? 8 : 0);
#pragma unroll
      for (int i = 0; i < 8; ++i) pa[i] = pp[i];
#pragma unroll
      for (int i = 0; i < 8; ++i) pa[8 + i] = pp[16 + i];
    }
    __syncthreads();

    // O += P(16x32) @ V(32x64); Vh is [dk][s] so B-fragments are contiguous
    o0 = wmma_f16(pa, b_frag_f16(Vh, SS, 0, kk, lane), o0);
    o1 = wmma_f16(pa, b_frag_f16(Vh, SS, 16, kk, lane), o1);
    o2 = wmma_f16(pa, b_frag_f16(Vh, SS, 32, kk, lane), o2);
    o3 = wmma_f16(pa, b_frag_f16(Vh, SS, 48, kk, lane), o3);
  }

  // normalize and store context in [b,s,h*dk] (= [B,S,D]) as f16
#pragma unroll
  for (int r = 0; r < 8; ++r) {
    int qrow = q0 + r + (hi ? 8 : 0);
    size_t base = ((size_t)b * SS + qrow) * DD + h * DKK + l;
    float inv = 1.0f / l_r[r];
    Ctx[base]      = (_Float16)(o0[r] * inv);
    Ctx[base + 16] = (_Float16)(o1[r] * inv);
    Ctx[base + 32] = (_Float16)(o2[r] * inv);
    Ctx[base + 48] = (_Float16)(o3[r] * inv);
  }
}

// -------------------- kernel 3: output projection -------------------------
__global__ void __launch_bounds__(256)
oproj_kernel(const _Float16* __restrict__ Ctx, const float* __restrict__ Wo,
             float* __restrict__ Out) {
  int wave = blockIdx.x * (blockDim.x >> 5) + (threadIdx.x >> 5);
  int lane = threadIdx.x & 31;
  int tm = wave >> 6, tn = wave & 63;
  int m0 = tm * 16, n0 = tn * 16;

  f8 acc = (f8)0.0f;
  for (int k = 0; k < DD; k += 32)
    acc = wmma_f16(a_frag_f16(Ctx, DD, m0, k, lane), b_frag_f32(Wo, DD, n0, k, lane), acc);

  int l = lane & 15, hi = lane >> 4;
#pragma unroll
  for (int r = 0; r < 8; ++r)
    Out[(size_t)(m0 + r + (hi ? 8 : 0)) * DD + n0 + l] = acc[r];
}

extern "C" void kernel_launch(void* const* d_in, const int* in_sizes, int n_in,
                              void* d_out, int out_size, void* d_ws, size_t ws_size,
                              hipStream_t stream) {
  const float* Q  = (const float*)d_in[0];
  const float* K  = (const float*)d_in[1];
  const float* V  = (const float*)d_in[2];
  const int* mask = (const int*)d_in[3];
  const float* Wq = (const float*)d_in[4];
  const float* Wk = (const float*)d_in[5];
  const float* Wv = (const float*)d_in[6];
  const float* Wo = (const float*)d_in[7];

  // workspace: 4 x 8MB f16 buffers (needs >= 32MB)
  char* ws = (char*)d_ws;
  _Float16* Qp  = (_Float16*)(ws);
  _Float16* Kp  = (_Float16*)(ws + ((size_t)8 << 20));
  _Float16* Vt  = (_Float16*)(ws + ((size_t)16 << 20));
  _Float16* Ctx = (_Float16*)(ws + ((size_t)24 << 20));

  // 256 M-tiles * 64 N-tiles = 16384 waves, 8 waves/block
  dim3 gproj(2048, 3);
  proj_kernel<<<gproj, 256, 0, stream>>>(Q, K, V, Wq, Wk, Wv, Qp, Kp, Vt);

  // B*H*(S/16) = 4096 single-wave blocks
  attn_kernel<<<4096, 32, 0, stream>>>(Qp, Kp, Vt, mask, Ctx);

  oproj_kernel<<<2048, 256, 0, stream>>>(Ctx, Wo, (float*)d_out);
}
